// PointerNet_30949534335591
// MI455X (gfx1250) — compile-verified, matrix operand order checked by reading, not run on hardware
//
#include <hip/hip_runtime.h>

// Problem sizes (match reference)
#define Bq 256   // batch
#define Sq 512   // seq len
#define Iq 128   // input size
#define Hq 256   // hidden size
#define NWG 16   // persistent workgroups; WG wg owns hidden cols [wg*16, wg*16+16)

typedef __attribute__((ext_vector_type(16))) __bf16 v16bf;
typedef __attribute__((ext_vector_type(8)))  float  v8f;
typedef __attribute__((ext_vector_type(4)))  int    v4i;

struct Frag32 { uint4 lo; uint4 hi; };   // 32 bytes == v16bf

// Hardware f32 -> bf16 conversion (v_cvt; backend packs pairs where possible)
__device__ __forceinline__ unsigned short f2bf(float f) {
  return __builtin_bit_cast(unsigned short, (__bf16)f);
}
__device__ __forceinline__ float sigm(float x) { return 1.0f / (1.0f + __expf(-x)); }
__device__ __forceinline__ float tanh_f(float x) {
  float e = __expf(-2.0f * fabsf(x));
  float r = (1.0f - e) / (1.0f + e);
  return copysignf(r, x);
}

// Load a 16x32 bf16 WMMA operand fragment from LDS (row-major rows, 16B chunks
// per the CDNA5 16-bit A/B VGPR layout: K = [k0+koff, +8) and [k0+16+koff, +8))
__device__ __forceinline__ v16bf ldfrag(const unsigned short* row, int k0, int koff) {
  Frag32 t;
  t.lo = *(const uint4*)(row + k0 + koff);
  t.hi = *(const uint4*)(row + k0 + 16 + koff);
  return __builtin_bit_cast(v16bf, t);
}

__device__ __forceinline__ v8f wmma_bf16(v16bf a, v16bf b, v8f c) {
  return __builtin_amdgcn_wmma_f32_16x16x32_bf16(false, a, false, b, (short)0, c, false, false);
}

// Stage one 512B row of h from L2 into LDS. Prefer the CDNA5 async-to-LDS path
// (GLOBAL_LOAD_ASYNC_TO_LDS_B128, tracked by ASYNCcnt): no VGPR round-trip, and
// it frees LOADcnt/DScnt slots for the WMMA fragment pipeline.
__device__ __forceinline__ void stage_h_row(const unsigned short* __restrict__ src,
                                            unsigned short* __restrict__ dst) {
#if __has_builtin(__builtin_amdgcn_global_load_async_to_lds_b128)
  #pragma unroll
  for (int c = 0; c < Hq; c += 8) {
    __builtin_amdgcn_global_load_async_to_lds_b128(
        (v4i*)(src + c), (v4i*)(dst + c), /*offset=*/0, /*cpol=*/0);
  }
  asm volatile("s_wait_asynccnt 0x0" ::: "memory");
#else
  const uint4* s4 = (const uint4*)src;
  uint4* d4 = (uint4*)dst;
  #pragma unroll 8
  for (int c = 0; c < Hq / 8; ++c) d4[c] = s4[c];
#endif
}

// Zero output (ptr indices are provably all 0), h ping-pong buffer, barrier.
extern "C" __global__ void ptrnet_init(int* out, int n_out,
                                       unsigned short* hb, int n_h, unsigned* bar) {
  int i = blockIdx.x * blockDim.x + threadIdx.x;
  if (i < n_out) out[i] = 0;
  if (i < n_h)  hb[i] = 0;
  if (i < 2)    bar[i] = 0u;
}

// Sense-free grid barrier across NWG workgroups via device-scope atomics in d_ws.
__device__ __forceinline__ void grid_barrier(unsigned* bar) {
  __syncthreads();
  if (threadIdx.x == 0) {
    __threadfence();
    unsigned my   = __hip_atomic_load(&bar[1], __ATOMIC_RELAXED, __HIP_MEMORY_SCOPE_AGENT);
    unsigned prev = __hip_atomic_fetch_add(&bar[0], 1u, __ATOMIC_ACQ_REL, __HIP_MEMORY_SCOPE_AGENT);
    if (prev == NWG - 1u) {
      __hip_atomic_store(&bar[0], 0u, __ATOMIC_RELAXED, __HIP_MEMORY_SCOPE_AGENT);
      __hip_atomic_fetch_add(&bar[1], 1u, __ATOMIC_ACQ_REL, __HIP_MEMORY_SCOPE_AGENT);
    } else {
      while (__hip_atomic_load(&bar[1], __ATOMIC_ACQUIRE, __HIP_MEMORY_SCOPE_AGENT) == my)
        __builtin_amdgcn_s_sleep(1);
    }
    __threadfence();
  }
  __syncthreads();
}

// Persistent LSTM recurrence. 16 WGs x 256 threads (8 wave32).
// WG wg owns hidden columns j0 = wg*16 .. +16 => gate rows {g*H + j0 + c}.
// Wave w owns batch rows 32w..32w+31 of that strip (2 M-tiles x 4 gate N-tiles).
extern "C" __global__ __launch_bounds__(256, 1)
void ptrnet_persistent(const float* __restrict__ inputs,
                       const float* __restrict__ eWih, const float* __restrict__ eWhh,
                       const float* __restrict__ eBih, const float* __restrict__ eBhh,
                       const float* __restrict__ dWih, const float* __restrict__ dWhh,
                       const float* __restrict__ dBih, const float* __restrict__ dBhh,
                       unsigned short* __restrict__ hglob,   // [2][B][H] bf16 ping-pong
                       unsigned* __restrict__ bar) {
  extern __shared__ char smem[];
  unsigned short* xs  = (unsigned short*)(smem);                            // [256][128] bf16 (64KB)
  unsigned short* hs  = (unsigned short*)(smem + 65536);                    // [256][256] bf16 (128KB)
  unsigned short* wis = (unsigned short*)(smem + 65536 + 131072);           // [4*16][128] Wih slice (16KB)
  unsigned short* whs = (unsigned short*)(smem + 65536 + 131072 + 16384);   // [4*16][256] Whh slice (32KB)
  float*          bs  = (float*)(smem + 65536 + 131072 + 16384 + 32768);    // [4*16] bias (256B)

  const int tid  = threadIdx.x;
  const int wg   = blockIdx.x;
  const int w    = tid >> 5;        // wave 0..7
  const int l    = tid & 31;
  const int ln   = l & 15;
  const int koff = (l >> 4) << 3;   // 0 (lanes 0-15) / 8 (lanes 16-31)
  const int j0   = wg * 16;

  // ---- stage encoder weight slices as bf16 into LDS -------------------------
  for (int idx = tid; idx < 64 * Iq; idx += 256) {
    int r = idx >> 7, c = idx & (Iq - 1);
    int g = r >> 4, cl = r & 15;
    wis[idx] = f2bf(eWih[(size_t)(g * Hq + j0 + cl) * Iq + c]);
  }
  for (int idx = tid; idx < 64 * Hq; idx += 256) {
    int r = idx >> 8, c = idx & (Hq - 1);
    int g = r >> 4, cl = r & 15;
    whs[idx] = f2bf(eWhh[(size_t)(g * Hq + j0 + cl) * Hq + c]);
  }
  if (tid < 64) {
    int g = tid >> 4, cl = tid & 15;
    bs[tid] = eBih[g * Hq + j0 + cl] + eBhh[g * Hq + j0 + cl];
  }

  float creg[2][8];                 // cell state, persistent across all 1024 steps
  #pragma unroll
  for (int mt = 0; mt < 2; ++mt)
    #pragma unroll
    for (int r = 0; r < 8; ++r) creg[mt][r] = 0.0f;

  int p = 0;
  __syncthreads();

  // ============================ ENCODER ======================================
  for (int t = 0; t < Sq; ++t) {
    // stage x_t (one row per thread), fp32 -> bf16 via hardware cvt
    {
      const float* xr = inputs + ((size_t)tid * Sq + t) * Iq;
      __builtin_prefetch(xr + Iq, 0, 1);            // next timestep's row
      #pragma unroll 4
      for (int c = 0; c < Iq; c += 4) {
        float4 v = *(const float4*)(xr + c);
        *(unsigned*)(xs + tid * Iq + c)     = (unsigned)f2bf(v.x) | ((unsigned)f2bf(v.y) << 16);
        *(unsigned*)(xs + tid * Iq + c + 2) = (unsigned)f2bf(v.z) | ((unsigned)f2bf(v.w) << 16);
      }
    }
    // stage full h_{t-1} broadcast from L2 (one row per thread, async -> LDS)
    stage_h_row(hglob + (size_t)p * Bq * Hq + (size_t)tid * Hq, hs + tid * Hq);
    __syncthreads();

    v8f acc[2][4];
    #pragma unroll
    for (int mt = 0; mt < 2; ++mt)
      #pragma unroll
      for (int g = 0; g < 4; ++g) acc[mt][g] = (v8f){};

    // x_t @ Wih^T  (K = 128)
    const unsigned short* aX0 = xs + (32 * w + ln) * Iq;
    const unsigned short* aX1 = xs + (32 * w + 16 + ln) * Iq;
    #pragma unroll
    for (int kk = 0; kk < Iq; kk += 32) {
      v16bf a0 = ldfrag(aX0, kk, koff);
      v16bf a1 = ldfrag(aX1, kk, koff);
      #pragma unroll
      for (int g = 0; g < 4; ++g) {
        v16bf b = ldfrag(wis + (g * 16 + ln) * Iq, kk, koff);
        acc[0][g] = wmma_bf16(a0, b, acc[0][g]);
        acc[1][g] = wmma_bf16(a1, b, acc[1][g]);
      }
    }
    // h_{t-1} @ Whh^T  (K = 256)
    const unsigned short* aH0 = hs + (32 * w + ln) * Hq;
    const unsigned short* aH1 = hs + (32 * w + 16 + ln) * Hq;
    #pragma unroll
    for (int kk = 0; kk < Hq; kk += 32) {
      v16bf a0 = ldfrag(aH0, kk, koff);
      v16bf a1 = ldfrag(aH1, kk, koff);
      #pragma unroll
      for (int g = 0; g < 4; ++g) {
        v16bf b = ldfrag(whs + (g * 16 + ln) * Hq, kk, koff);
        acc[0][g] = wmma_bf16(a0, b, acc[0][g]);
        acc[1][g] = wmma_bf16(a1, b, acc[1][g]);
      }
    }

    // gates -> c,h ; publish h slice to L2 (bf16)
    float bi = bs[ln], bf = bs[16 + ln], bg = bs[32 + ln], bo = bs[48 + ln];
    unsigned short* hnext = hglob + (size_t)(p ^ 1) * Bq * Hq;
    #pragma unroll
    for (int mt = 0; mt < 2; ++mt)
      #pragma unroll
      for (int r = 0; r < 8; ++r) {
        float iv = sigm(acc[mt][0][r] + bi);
        float fv = sigm(acc[mt][1][r] + bf);
        float gv = tanh_f(acc[mt][2][r] + bg);
        float ov = sigm(acc[mt][3][r] + bo);
        float cv = fv * creg[mt][r] + iv * gv;
        creg[mt][r] = cv;
        int row = 32 * w + 16 * mt + r + koff;      // C-layout: lanes>=16 -> M+8
        hnext[(size_t)row * Hq + j0 + ln] = f2bf(ov * tanh_f(cv));
      }

    grid_barrier(bar);
    p ^= 1;
  }

  // ============================ DECODER ======================================
  // ptr = argmax over [B,1] logits == 0 always => decoder input is constant
  // x0 = inputs[:,0,:] (zero on step 0). Hoist x0 @ Wih^T once.
  __syncthreads();
  for (int idx = tid; idx < 64 * Iq; idx += 256) {
    int r = idx >> 7, c = idx & (Iq - 1);
    int g = r >> 4, cl = r & 15;
    wis[idx] = f2bf(dWih[(size_t)(g * Hq + j0 + cl) * Iq + c]);
  }
  for (int idx = tid; idx < 64 * Hq; idx += 256) {
    int r = idx >> 8, c = idx & (Hq - 1);
    int g = r >> 4, cl = r & 15;
    whs[idx] = f2bf(dWhh[(size_t)(g * Hq + j0 + cl) * Hq + c]);
  }
  if (tid < 64) {
    int g = tid >> 4, cl = tid & 15;
    bs[tid] = dBih[g * Hq + j0 + cl] + dBhh[g * Hq + j0 + cl];
  }
  {
    const float* xr = inputs + (size_t)tid * Sq * Iq;   // inputs[b, 0, :]
    #pragma unroll 4
    for (int c = 0; c < Iq; c += 4) {
      float4 v = *(const float4*)(xr + c);
      *(unsigned*)(xs + tid * Iq + c)     = (unsigned)f2bf(v.x) | ((unsigned)f2bf(v.y) << 16);
      *(unsigned*)(xs + tid * Iq + c + 2) = (unsigned)f2bf(v.z) | ((unsigned)f2bf(v.w) << 16);
    }
  }
  __syncthreads();

  v8f xw[2][4];
  #pragma unroll
  for (int mt = 0; mt < 2; ++mt)
    #pragma unroll
    for (int g = 0; g < 4; ++g) xw[mt][g] = (v8f){};
  {
    const unsigned short* aX0 = xs + (32 * w + ln) * Iq;
    const unsigned short* aX1 = xs + (32 * w + 16 + ln) * Iq;
    #pragma unroll
    for (int kk = 0; kk < Iq; kk += 32) {
      v16bf a0 = ldfrag(aX0, kk, koff);
      v16bf a1 = ldfrag(aX1, kk, koff);
      #pragma unroll
      for (int g = 0; g < 4; ++g) {
        v16bf b = ldfrag(wis + (g * 16 + ln) * Iq, kk, koff);
        xw[0][g] = wmma_bf16(a0, b, xw[0][g]);
        xw[1][g] = wmma_bf16(a1, b, xw[1][g]);
      }
    }
  }

  for (int d = 0; d < Sq; ++d) {
    stage_h_row(hglob + (size_t)p * Bq * Hq + (size_t)tid * Hq, hs + tid * Hq);
    __syncthreads();

    v8f acc[2][4];
    if (d == 0) {
      #pragma unroll
      for (int mt = 0; mt < 2; ++mt)
        #pragma unroll
        for (int g = 0; g < 4; ++g) acc[mt][g] = (v8f){};
    } else {
      #pragma unroll
      for (int mt = 0; mt < 2; ++mt)
        #pragma unroll
        for (int g = 0; g < 4; ++g) acc[mt][g] = xw[mt][g];
    }

    const unsigned short* aH0 = hs + (32 * w + ln) * Hq;
    const unsigned short* aH1 = hs + (32 * w + 16 + ln) * Hq;
    #pragma unroll
    for (int kk = 0; kk < Hq; kk += 32) {
      v16bf a0 = ldfrag(aH0, kk, koff);
      v16bf a1 = ldfrag(aH1, kk, koff);
      #pragma unroll
      for (int g = 0; g < 4; ++g) {
        v16bf b = ldfrag(whs + (g * 16 + ln) * Hq, kk, koff);
        acc[0][g] = wmma_bf16(a0, b, acc[0][g]);
        acc[1][g] = wmma_bf16(a1, b, acc[1][g]);
      }
    }

    float bi = bs[ln], bf = bs[16 + ln], bg = bs[32 + ln], bo = bs[48 + ln];
    unsigned short* hnext = hglob + (size_t)(p ^ 1) * Bq * Hq;
    #pragma unroll
    for (int mt = 0; mt < 2; ++mt)
      #pragma unroll
      for (int r = 0; r < 8; ++r) {
        float iv = sigm(acc[mt][0][r] + bi);
        float fv = sigm(acc[mt][1][r] + bf);
        float gv = tanh_f(acc[mt][2][r] + bg);
        float ov = sigm(acc[mt][3][r] + bo);
        float cv = fv * creg[mt][r] + iv * gv;
        creg[mt][r] = cv;
        int row = 32 * w + 16 * mt + r + koff;
        hnext[(size_t)row * Hq + j0 + ln] = f2bf(ov * tanh_f(cv));
      }

    grid_barrier(bar);
    p ^= 1;
  }
  // Output (all-zero pointer indices) is written by ptrnet_init.
}

extern "C" void kernel_launch(void* const* d_in, const int* in_sizes, int n_in,
                              void* d_out, int out_size, void* d_ws, size_t ws_size,
                              hipStream_t stream) {
  (void)in_sizes; (void)n_in; (void)ws_size;
  const float* inputs = (const float*)d_in[0];
  const float* eWih   = (const float*)d_in[1];
  const float* eWhh   = (const float*)d_in[2];
  const float* eBih   = (const float*)d_in[3];
  const float* eBhh   = (const float*)d_in[4];
  const float* dWih   = (const float*)d_in[5];
  const float* dWhh   = (const float*)d_in[6];
  const float* dBih   = (const float*)d_in[7];
  const float* dBhh   = (const float*)d_in[8];
  // d_in[9] (ptr_W) and d_in[10] (ptr_b) provably do not affect the output:
  // argmax over a single-column logit is always 0.

  unsigned* bar = (unsigned*)d_ws;
  unsigned short* hglob = (unsigned short*)((char*)d_ws + 256);  // 2*B*H bf16 = 256KB

  const int n_h = 2 * Bq * Hq;
  int init_n = out_size > n_h ? out_size : n_h;
  ptrnet_init<<<(init_n + 255) / 256, 256, 0, stream>>>((int*)d_out, out_size, hglob, n_h, bar);

  size_t smem = 65536 + 131072 + 16384 + 32768 + 256;  // ~240.25 KB of the 320KB WGP LDS
  (void)hipFuncSetAttribute((const void*)ptrnet_persistent,
                            hipFuncAttributeMaxDynamicSharedMemorySize, (int)smem);
  ptrnet_persistent<<<NWG, 256, smem, stream>>>(inputs, eWih, eWhh, eBih, eBhh,
                                                dWih, dWhh, dBih, dBhh, hglob, bar);
}
